// Model_55705725829413
// MI455X (gfx1250) — matrix operand order repaired and checked by reading, not compiled, not run on hardware
//
#include <hip/hip_runtime.h>
#include <hip/hip_bf16.h>
#include <stdint.h>
#include <stddef.h>

typedef __attribute__((ext_vector_type(16))) __bf16 v16bf;
typedef __attribute__((ext_vector_type(8)))  float  v8f;

#define NUM_TYPES 10
#define NUM_RELS  18
#define H_DIM     128
#define OUT_DIM   64
#define DIM_IN    768

__constant__ int c_rel_dst[NUM_RELS] = {1,0,2,0,3,2,4,2,7,0,8,1,5,3,6,2,9,2};

static const int h_counts[NUM_TYPES] = {8000,20000,20000,20000,2000,5000,500,5000,10000,20000};
static const int h_rel_src[NUM_RELS] = {0,1,0,2,2,3,2,4,0,7,1,8,3,5,2,6,2,9};
static const int h_rel_dst[NUM_RELS] = {1,0,2,0,3,2,4,2,7,0,8,1,5,3,6,2,9,2};

__device__ __forceinline__ unsigned short f2bf(float f) {
  unsigned int u = __float_as_uint(f);
  u += 0x7FFFu + ((u >> 16) & 1u);   // round-to-nearest-even truncation to bf16
  return (unsigned short)(u >> 16);
}

// ---------------------------------------------------------------------------
// WMMA GEMM: C[M,N] (+)= rowscale(A[M,K]) @ W[K,N] (+ bias) (relu?)
// A, W fp32 in global; converted to bf16 while staging into LDS.
// Block: 128 threads = 4 waves. Block tile: 64 rows x N cols. K chunk: 32.
// N = NT*16 (NT = 4 -> 64, NT = 8 -> 128). K must be a multiple of 32.
// ---------------------------------------------------------------------------
template<int NT>
__global__ __launch_bounds__(128)
void gemm_wmma(const float* __restrict__ A, int lda,
               const float* __restrict__ W,
               const float* __restrict__ bias,
               float* __restrict__ C,
               int M, int K,
               const float* __restrict__ rscale,
               int accum, int relu)
{
  constexpr int N = NT * 16;
  __shared__ unsigned short Atile[64][34];     // [row][k], padded
  __shared__ unsigned short BtileT[N][34];     // [col][k], transposed, padded

  const int tid     = threadIdx.x;
  const int lane    = tid & 31;
  const int wave    = tid >> 5;
  const int lane15  = lane & 15;
  const int kh      = lane >> 4;               // half-wave selector
  const int blockRow = blockIdx.x * 64;

  v8f acc[NT];
  #pragma unroll
  for (int i = 0; i < NT; ++i)
    #pragma unroll
    for (int j = 0; j < 8; ++j) acc[i][j] = 0.0f;

  // A-tile load mapping: 2 threads per row, 16 consecutive cols each
  const int arow = tid >> 1;
  const int acb  = (tid & 1) * 16;
  const int grow = blockRow + arow;
  float ascale = 1.0f;
  if (rscale != nullptr && grow < M) ascale = 1.0f / fmaxf(rscale[grow], 1.0f);

  for (int kc = 0; kc < K; kc += 32) {
    // ---- stage A (64x32) as bf16 ----
    if (grow < M) {
      const float4* srcp = (const float4*)(A + (size_t)grow * lda + kc + acb);
      #pragma unroll
      for (int j = 0; j < 4; ++j) {
        float4 v = srcp[j];
        v.x *= ascale; v.y *= ascale; v.z *= ascale; v.w *= ascale;
        unsigned int p0 = (unsigned int)f2bf(v.x) | ((unsigned int)f2bf(v.y) << 16);
        unsigned int p1 = (unsigned int)f2bf(v.z) | ((unsigned int)f2bf(v.w) << 16);
        *(unsigned int*)&Atile[arow][acb + 4*j]     = p0;
        *(unsigned int*)&Atile[arow][acb + 4*j + 2] = p1;
      }
    } else {
      #pragma unroll
      for (int j = 0; j < 4; ++j) {
        *(unsigned int*)&Atile[arow][acb + 4*j]     = 0u;
        *(unsigned int*)&Atile[arow][acb + 4*j + 2] = 0u;
      }
    }
    // ---- stage B (32xN) transposed as bf16 ----
    #pragma unroll
    for (int i = 0; i < NT; ++i) {
      int chunk = tid + i * 128;
      int row = chunk / (N / 4);
      int col = (chunk % (N / 4)) * 4;
      float4 v = *(const float4*)(W + (size_t)(kc + row) * N + col);
      BtileT[col + 0][row] = f2bf(v.x);
      BtileT[col + 1][row] = f2bf(v.y);
      BtileT[col + 2][row] = f2bf(v.z);
      BtileT[col + 3][row] = f2bf(v.w);
    }
    __syncthreads();

    // ---- A fragment (16x32 bf16, ISA layout) ----
    union { v16bf v; unsigned int u[8]; } af;
    {
      const int m = wave * 16 + lane15;
      #pragma unroll
      for (int vi = 0; vi < 8; ++vi) {
        int kbase = (vi < 4) ? (kh * 8 + 2 * vi) : (16 + kh * 8 + 2 * (vi - 4));
        af.u[vi] = *(const unsigned int*)&Atile[m][kbase];
      }
    }
    // ---- loop over 16-wide column tiles ----
    #pragma unroll
    for (int nt = 0; nt < NT; ++nt) {
      union { v16bf v; unsigned int u[8]; } bf;
      const int n = nt * 16 + lane15;
      #pragma unroll
      for (int vi = 0; vi < 8; ++vi) {
        int k = (vi < 4) ? (kh * 8 + 2 * vi) : (16 + kh * 8 + 2 * (vi - 4));
        bf.u[vi] = *(const unsigned int*)&BtileT[n][k];
      }
      acc[nt] = __builtin_amdgcn_wmma_f32_16x16x32_bf16(
          false, af.v, false, bf.v, (short)0, acc[nt], false, false);
    }
    __syncthreads();
  }

  // ---- epilogue: C/D layout lane(n=lane15, mh=lane>>4), VGPR v -> row mh*8+v
  const int mh = kh;
  #pragma unroll
  for (int nt = 0; nt < NT; ++nt) {
    const int n = nt * 16 + lane15;
    const float b = bias ? bias[n] : 0.0f;
    #pragma unroll
    for (int vi = 0; vi < 8; ++vi) {
      const int row = blockRow + wave * 16 + mh * 8 + vi;
      if (row < M) {
        float val = acc[nt][vi] + b;
        size_t idx = (size_t)row * N + n;
        if (accum) val += C[idx];
        if (relu)  val = fmaxf(val, 0.0f);
        C[idx] = val;
      }
    }
  }
}

// ---------------------------------------------------------------------------
// Edge-parallel scatter for segment mean: msum[dst] += h[src], cnt[dst] += 1
// 32 lanes per edge (128 channels as 32 float4 chunks), f32 global atomics.
// ---------------------------------------------------------------------------
__global__ void scatter_mean_accum(const float* __restrict__ hsrc,
                                   const int* __restrict__ srcIdx,
                                   const int* __restrict__ dstIdx,
                                   float* __restrict__ msum,
                                   float* __restrict__ cnt,
                                   int E)
{
  int tid = blockIdx.x * blockDim.x + threadIdx.x;
  int e   = tid >> 5;
  int c4  = (tid & 31) << 2;
  if (e >= E) return;
  int s = srcIdx[e];
  int d = dstIdx[e];
  float4 v = *(const float4*)(hsrc + (size_t)s * H_DIM + c4);
  float* p = msum + (size_t)d * H_DIM + c4;
  atomicAdd(p + 0, v.x);
  atomicAdd(p + 1, v.y);
  atomicAdd(p + 2, v.z);
  atomicAdd(p + 3, v.w);
  if (c4 == 0) atomicAdd(cnt + d, 1.0f);
}

__global__ void zero2_kernel(float* a, size_t na, float* b, size_t nb)
{
  size_t stride = (size_t)gridDim.x * blockDim.x;
  size_t total = na + nb;
  for (size_t i = (size_t)blockIdx.x * blockDim.x + threadIdx.x; i < total; i += stride) {
    if (i < na) a[i] = 0.0f; else b[i - na] = 0.0f;
  }
}

__global__ void relu_kernel(float* __restrict__ p, size_t n)
{
  size_t stride = (size_t)gridDim.x * blockDim.x;
  for (size_t i = (size_t)blockIdx.x * blockDim.x + threadIdx.x; i < n; i += stride)
    p[i] = fmaxf(p[i], 0.0f);
}

// WrC[t] = sum_{r: dst(r)=t} Wr[r];  bC[t] = sum bl[r]
__global__ void combine_wr(const float* __restrict__ Wr, const float* __restrict__ bl,
                           float* __restrict__ WrC, float* __restrict__ bC,
                           int din, int dout)
{
  int t = blockIdx.y;
  int i = blockIdx.x * blockDim.x + threadIdx.x;
  int dd = din * dout;
  if (i < dd) {
    float acc = 0.0f;
    for (int r = 0; r < NUM_RELS; ++r)
      if (c_rel_dst[r] == t) acc += Wr[(size_t)r * dd + i];
    WrC[(size_t)t * dd + i] = acc;
  }
  if (i < dout) {
    float b = 0.0f;
    for (int r = 0; r < NUM_RELS; ++r)
      if (c_rel_dst[r] == t) b += bl[(size_t)r * dout + i];
    bC[(size_t)t * dout + i] = b;
  }
}

__global__ void classify_kernel(const float* __restrict__ hd, const float* __restrict__ hs,
                                const int* __restrict__ eli, float* __restrict__ out,
                                int n)
{
  int i = blockIdx.x * blockDim.x + threadIdx.x;
  if (i >= n) return;
  int a = eli[i];
  int b = eli[n + i];
  const float4* pa = (const float4*)(hd + (size_t)a * OUT_DIM);
  const float4* pb = (const float4*)(hs + (size_t)b * OUT_DIM);
  float acc = 0.0f;
  #pragma unroll
  for (int c = 0; c < OUT_DIM / 4; ++c) {
    float4 xv = pa[c]; float4 yv = pb[c];
    acc += xv.x*yv.x + xv.y*yv.y + xv.z*yv.z + xv.w*yv.w;
  }
  out[i] = acc;
}

static inline void launch_gemm(const float* A, int lda, const float* W, const float* bias,
                               float* C, int M, int K, int N, const float* rscale,
                               int accum, int relu, hipStream_t s)
{
  dim3 grid((M + 63) / 64);
  if (N == 128)
    gemm_wmma<8><<<grid, 128, 0, s>>>(A, lda, W, bias, C, M, K, rscale, accum, relu);
  else
    gemm_wmma<4><<<grid, 128, 0, s>>>(A, lda, W, bias, C, M, K, rscale, accum, relu);
}

extern "C" void kernel_launch(void* const* d_in, const int* in_sizes, int n_in,
                              void* d_out, int out_size, void* d_ws, size_t ws_size,
                              hipStream_t stream)
{
  (void)n_in; (void)out_size; (void)ws_size;

  const float* x[NUM_TYPES];
  for (int t = 0; t < NUM_TYPES; ++t) x[t] = (const float*)d_in[t];
  const float* Win  = (const float*)d_in[10];
  const float* b_in = (const float*)d_in[11];
  const float* Wl[3] = {(const float*)d_in[12], (const float*)d_in[15], (const float*)d_in[18]};
  const float* bl[3] = {(const float*)d_in[13], (const float*)d_in[16], (const float*)d_in[19]};
  const float* Wr[3] = {(const float*)d_in[14], (const float*)d_in[17], (const float*)d_in[20]};
  const int* edge_index = (const int*)d_in[21];
  const int* eli        = (const int*)d_in[22];
  const int E  = in_sizes[21] / (NUM_RELS * 2);
  const int NL = in_sizes[22] / 2;

  // per-type feature offsets (elements)
  size_t off128[NUM_TYPES], off64[NUM_TYPES], totalN = 0;
  for (int t = 0; t < NUM_TYPES; ++t) {
    off128[t] = totalN * 128;
    off64[t]  = totalN * 64;
    totalN   += (size_t)h_counts[t];
  }

  // workspace carve-out (~124 MB)
  char* ws = (char*)d_ws;
  float* hA   = (float*)ws; ws += totalN * 128 * sizeof(float);
  float* hB   = (float*)ws; ws += totalN * 128 * sizeof(float);
  float* msum = (float*)ws; ws += (size_t)20000 * 128 * sizeof(float);
  float* cnt  = (float*)ws; ws += (size_t)20000 * sizeof(float);
  float* WrC  = (float*)ws; ws += (size_t)NUM_TYPES * 128 * 128 * sizeof(float);
  float* bC   = (float*)ws; ws += (size_t)NUM_TYPES * 128 * sizeof(float);

  // 1) input projection: hA[t] = x[t] @ Win[t] + b_in[t]
  for (int t = 0; t < NUM_TYPES; ++t)
    launch_gemm(x[t], DIM_IN, Win + (size_t)t * DIM_IN * H_DIM,
                b_in + (size_t)t * H_DIM, hA + off128[t],
                h_counts[t], DIM_IN, H_DIM, nullptr, 0, 0, stream);

  float* hin = hA;
  float* hout = hB;
  for (int l = 0; l < 3; ++l) {
    const int din  = H_DIM;
    const int dout = (l == 2) ? OUT_DIM : H_DIM;
    const size_t* offo = (l == 2) ? off64 : off128;

    // combined self-weights: WrC[t] = sum_r Wr[r], bC[t] = sum_r bl[r]
    {
      dim3 grid((din * dout + 255) / 256, NUM_TYPES);
      combine_wr<<<grid, 256, 0, stream>>>(Wr[l], bl[l], WrC, bC, din, dout);
    }
    // self term: hout[t] = hin[t] @ WrC[t] + bC[t]
    for (int t = 0; t < NUM_TYPES; ++t)
      launch_gemm(hin + off128[t], din, WrC + (size_t)t * din * dout,
                  bC + (size_t)t * dout, hout + offo[t],
                  h_counts[t], din, dout, nullptr, 0, 0, stream);

    // relation terms: hout[d] += mean_r @ Wl[r]
    for (int r = 0; r < NUM_RELS; ++r) {
      const int s = h_rel_src[r], d = h_rel_dst[r];
      const int nd = h_counts[d];
      const size_t nm = (size_t)nd * din;
      {
        int blocks = (int)((nm + (size_t)nd + 255) / 256);
        if (blocks > 4096) blocks = 4096;
        zero2_kernel<<<blocks, 256, 0, stream>>>(msum, nm, cnt, (size_t)nd);
      }
      {
        const int* srcIdx = edge_index + (size_t)r * 2 * E;
        const int* dstIdx = srcIdx + E;
        dim3 grid((unsigned)(((size_t)E * 32 + 255) / 256));
        scatter_mean_accum<<<grid, 256, 0, stream>>>(hin + off128[s], srcIdx, dstIdx,
                                                     msum, cnt, E);
      }
      launch_gemm(msum, din, Wl[l] + (size_t)r * din * dout, nullptr,
                  hout + offo[d], nd, din, dout, cnt, 1, 0, stream);
    }

    if (l < 2)
      relu_kernel<<<2048, 256, 0, stream>>>(hout, totalN * 128);

    float* tmp = hin; hin = hout; hout = tmp;   // hin now holds this layer's output
  }

  // classifier: dot(h3_drug[eli0], h3_disorder[eli1]) over 64 dims
  {
    dim3 grid((NL + 255) / 256);
    classify_kernel<<<grid, 256, 0, stream>>>(hin + off64[0], hin + off64[1],
                                              eli, (float*)d_out, NL);
  }
}
